// AdaptiveAttention_50319836840405
// MI455X (gfx1250) — matrix operand order repaired and checked by reading, not compiled, not run on hardware
//
#include <hip/hip_runtime.h>
#include <hip/hip_bf16.h>

typedef __attribute__((ext_vector_type(16))) __bf16 v16bf;
typedef __attribute__((ext_vector_type(8)))  float  v8f;

#define HIDDEN 1024
#define ATT    512
#define BATCH  256
#define PATCH  196
#define MROWS  (BATCH * PATCH)   /* 50176, divisible by 16 */
#define KCH    32                /* K per bf16 WMMA */
#define NKC    (HIDDEN / KCH)    /* 32 K-chunks */
#define NSUB   (ATT / 16)        /* 32 N-subtiles */

__device__ __forceinline__ unsigned short f2bf_raw(float f) {
  unsigned u = __builtin_bit_cast(unsigned, f);
  unsigned r = u + 0x7FFFu + ((u >> 16) & 1u);   // round-to-nearest-even
  return (unsigned short)(r >> 16);
}

// Workgroup barrier that orders only LDS traffic (no global-memory drain).
// Each wave waits for its own outstanding DS ops, then uses the CDNA5
// split barrier.  Avoids the s_wait_loadcnt 0 that __syncthreads() emits.
__device__ __forceinline__ void lds_barrier() {
  asm volatile("s_wait_dscnt 0x0\n\t"
               "s_barrier_signal -1\n\t"
               "s_barrier_wait -1" ::: "memory");
}

// -------------------------------------------------------------------------
// Pack f32 weights (K=1024 x N=512, row-major) into bf16 WMMA-B fragment
// order: frag[(c*NSUB+j)*512 + lane*16 + slot]
//   lane = (n&15) + 16*((kk>>4)&1), slot = kk & 15   (kk = k & 31, c = k>>5)
// -------------------------------------------------------------------------
__global__ void pack_w_kernel(const float* __restrict__ w0,
                              const float* __restrict__ w1,
                              const float* __restrict__ w2,
                              unsigned short* __restrict__ p0,
                              unsigned short* __restrict__ p1,
                              unsigned short* __restrict__ p2) {
  int idx = blockIdx.x * blockDim.x + threadIdx.x;  // 0 .. 1024*512-1
  const float*    src = (blockIdx.y == 0) ? w0 : (blockIdx.y == 1) ? w1 : w2;
  unsigned short* dst = (blockIdx.y == 0) ? p0 : (blockIdx.y == 1) ? p1 : p2;
  int k = idx >> 9;
  int n = idx & 511;
  int c = k >> 5, kk = k & 31;
  int j = n >> 4, nn = n & 15;
  int lane = nn + (((kk >> 4) & 1) << 4);
  int slot = kk & 15;
  size_t off = ((size_t)(c * NSUB + j) * 32 + lane) * 16 + slot;
  dst[off] = f2bf_raw(src[idx]);
}

// -------------------------------------------------------------------------
// Big fused kernel: zt[row] = sum_n tanh( (spatial @ Wcnn)[row][n]
//                                         + dec[row/196][n] ) * watt[n]
// 16-row M-tile per workgroup, 8 wave32, 4 N-subtiles per wave.
// B register-double-buffered across K-chunks; A load hoisted one chunk
// ahead; in-loop barriers are DS-only split barriers.
// -------------------------------------------------------------------------
__global__ __launch_bounds__(256) void
zt_kernel(const float* __restrict__ A,            // [MROWS][1024]
          const unsigned short* __restrict__ Wp,  // packed Wcnn bf16
          const float* __restrict__ dec,          // [256][512]
          const float* __restrict__ watt,         // [512]
          float* __restrict__ zt)                 // [MROWS]
{
  __shared__ __attribute__((aligned(32))) __bf16 lds_a[2][512];
  __shared__ float zt_lds[16];

  const int t    = threadIdx.x;
  const int lane = t & 31;
  const int wave = t >> 5;
  const int row0 = blockIdx.x * 16;

  if (t < 16) zt_lds[t] = 0.0f;

  // A-staging mapping for this thread's element pair (e2, e2+1)
  const int e2     = t * 2;
  const int sm     = e2 >> 5;                         // row in tile
  const int sk     = e2 & 31;                         // even k in chunk
  const int s_lane = sm + ((sk >> 4) << 4);
  const int s_slot = sk & 15;                         // even
  const float* aRow = A + (size_t)(row0 + sm) * HIDDEN + sk;
  const unsigned short* bLane =
      Wp + (size_t)(wave * 4) * 512 + lane * 16;      // + j*512 + c*NSUB*512

  v8f acc[4];
#pragma unroll
  for (int j = 0; j < 4; ++j)
#pragma unroll
    for (int i = 0; i < 8; ++i) acc[j][i] = 0.0f;

  // prologue: A chunk 0 and B fragments for chunk 0
  float2 fcur = *(const float2*)(aRow);
  v16bf bcur[4];
#pragma unroll
  for (int j = 0; j < 4; ++j)
    bcur[j] = *(const v16bf*)(bLane + (size_t)j * 512);

#pragma unroll 2
  for (int c = 0; c < NKC; ++c) {
    // stage chunk c's A (already in registers) into LDS
    unsigned pk = (unsigned)f2bf_raw(fcur.x) | ((unsigned)f2bf_raw(fcur.y) << 16);
    *(unsigned*)&lds_a[c & 1][s_lane * 16 + s_slot] = pk;
    // issue next chunk's A load early (whole compute phase to complete)
    if (c + 1 < NKC) fcur = *(const float2*)(aRow + (c + 1) * KCH);
    lds_barrier();

    v16bf a = *(const v16bf*)&lds_a[c & 1][lane * 16];

    // issue next chunk's B loads before consuming current B
    v16bf bnext[4];
    if (c + 1 < NKC) {
      const unsigned short* wb = bLane + (size_t)(c + 1) * NSUB * 512;
#pragma unroll
      for (int j = 0; j < 4; ++j)
        bnext[j] = *(const v16bf*)(wb + (size_t)j * 512);
    }

#pragma unroll
    for (int j = 0; j < 4; ++j)
      acc[j] = __builtin_amdgcn_wmma_f32_16x16x32_bf16(
          false, a, false, bcur[j], (short)0, acc[j], false, false);

#pragma unroll
    for (int j = 0; j < 4; ++j) bcur[j] = bnext[j];
  }

  // epilogue: tanh + dot with watt, reduce N across lanes -> zt
  const int half = lane >> 4;
  const int nl   = lane & 15;
#pragma unroll
  for (int r = 0; r < 8; ++r) {
    int m   = r + 8 * half;                // C layout: lanes16-31 hold M=8+r
    int row = row0 + m;
    int b   = row / PATCH;
    float v = 0.0f;
#pragma unroll
    for (int j = 0; j < 4; ++j) {
      int n = wave * 64 + j * 16 + nl;     // C layout: N = lane&15 (+subtile)
      v += tanhf(acc[j][r] + dec[b * ATT + n]) * watt[n];
    }
    v += __shfl_xor(v, 1);
    v += __shfl_xor(v, 2);
    v += __shfl_xor(v, 4);
    v += __shfl_xor(v, 8);
    if (nl == 0) atomicAdd(&zt_lds[m], v);
  }
  lds_barrier();
  if (t < 16) zt[row0 + t] = zt_lds[t];
}

// -------------------------------------------------------------------------
// Small fused kernel: dec = decoder @ Wdec (stored f32),
//                     sen = tanh(dec + st @ Wsen), out_b = sen @ watt
// -------------------------------------------------------------------------
__global__ __launch_bounds__(256) void
decsen_kernel(const float* __restrict__ Adec,          // [256][1024]
              const float* __restrict__ Ast,           // [256][1024]
              const unsigned short* __restrict__ Wdp,
              const unsigned short* __restrict__ Wsp,
              const float* __restrict__ watt,
              float* __restrict__ dec_out,             // [256][512]
              float* __restrict__ out_b)               // [256]
{
  __shared__ __attribute__((aligned(32))) __bf16 lds_d[2][512];
  __shared__ __attribute__((aligned(32))) __bf16 lds_s[2][512];
  __shared__ float o_lds[16];

  const int t    = threadIdx.x;
  const int lane = t & 31;
  const int wave = t >> 5;
  const int row0 = blockIdx.x * 16;

  if (t < 16) o_lds[t] = 0.0f;

  const int e2     = t * 2;
  const int sm     = e2 >> 5;
  const int sk     = e2 & 31;
  const int s_lane = sm + ((sk >> 4) << 4);
  const int s_slot = sk & 15;
  const float* dRow = Adec + (size_t)(row0 + sm) * HIDDEN + sk;
  const float* sRow = Ast  + (size_t)(row0 + sm) * HIDDEN + sk;

  v8f accD[4], accS[4];
#pragma unroll
  for (int j = 0; j < 4; ++j)
#pragma unroll
    for (int i = 0; i < 8; ++i) { accD[j][i] = 0.0f; accS[j][i] = 0.0f; }

  for (int c = 0; c < NKC; ++c) {
    float2 fd = *(const float2*)(dRow + c * KCH);
    float2 fs = *(const float2*)(sRow + c * KCH);
    *(unsigned*)&lds_d[c & 1][s_lane * 16 + s_slot] =
        (unsigned)f2bf_raw(fd.x) | ((unsigned)f2bf_raw(fd.y) << 16);
    *(unsigned*)&lds_s[c & 1][s_lane * 16 + s_slot] =
        (unsigned)f2bf_raw(fs.x) | ((unsigned)f2bf_raw(fs.y) << 16);
    lds_barrier();

    v16bf ad = *(const v16bf*)&lds_d[c & 1][lane * 16];
    v16bf as = *(const v16bf*)&lds_s[c & 1][lane * 16];
    const size_t cb = (size_t)c * NSUB * 512;
#pragma unroll
    for (int j = 0; j < 4; ++j) {
      int sub = wave * 4 + j;
      v16bf bd = *(const v16bf*)(Wdp + cb + (size_t)sub * 512 + lane * 16);
      v16bf bs = *(const v16bf*)(Wsp + cb + (size_t)sub * 512 + lane * 16);
      accD[j] = __builtin_amdgcn_wmma_f32_16x16x32_bf16(
          false, ad, false, bd, (short)0, accD[j], false, false);
      accS[j] = __builtin_amdgcn_wmma_f32_16x16x32_bf16(
          false, as, false, bs, (short)0, accS[j], false, false);
    }
  }

  const int half = lane >> 4;
  const int nl   = lane & 15;
#pragma unroll
  for (int r = 0; r < 8; ++r) {
    int m   = r + 8 * half;
    int row = row0 + m;                    // row == batch index here
    float v = 0.0f;
#pragma unroll
    for (int j = 0; j < 4; ++j) {
      int   n  = wave * 64 + j * 16 + nl;
      float dv = accD[j][r];
      dec_out[(size_t)row * ATT + n] = dv;          // pre-tanh dec_out
      v += tanhf(dv + accS[j][r]) * watt[n];        // sen . watt
    }
    v += __shfl_xor(v, 1);
    v += __shfl_xor(v, 2);
    v += __shfl_xor(v, 4);
    v += __shfl_xor(v, 8);
    if (nl == 0) atomicAdd(&o_lds[m], v);
  }
  lds_barrier();
  if (t < 16) out_b[row0 + t] = o_lds[t];
}

// -------------------------------------------------------------------------
// Per-batch: softmax(zt) -> alpha; softmax([zt,out]) -> beta; ct; c_hat
// -------------------------------------------------------------------------
__global__ __launch_bounds__(256) void
final_kernel(const float* __restrict__ zt,     // [MROWS]
             const float* __restrict__ out_b,  // [256]
             const float* __restrict__ A,      // spatial [256][196][1024]
             const float* __restrict__ st,     // [256][1024]
             float* __restrict__ alpha_out,
             float* __restrict__ beta_out,
             float* __restrict__ chat_out)
{
  __shared__ float red[256];
  __shared__ float alpha_sh[PATCH];
  const int b = blockIdx.x;
  const int t = threadIdx.x;

  float z = (t < PATCH) ? zt[b * PATCH + t] : -__FLT_MAX__;
  red[t] = z;
  __syncthreads();
  for (int s = 128; s > 0; s >>= 1) {
    if (t < s) red[t] = fmaxf(red[t], red[t + s]);
    __syncthreads();
  }
  float m1 = red[0];
  __syncthreads();

  float e = (t < PATCH) ? __expf(z - m1) : 0.0f;
  red[t] = e;
  __syncthreads();
  for (int s = 128; s > 0; s >>= 1) {
    if (t < s) red[t] += red[t + s];
    __syncthreads();
  }
  float s1 = red[0];
  __syncthreads();

  float ob   = out_b[b];
  float m2   = fmaxf(m1, ob);
  float eo   = __expf(ob - m2);
  float s2   = s1 * __expf(m1 - m2) + eo;   // softmax over 197 entries
  float beta = eo / s2;

  if (t < PATCH) {
    float al = e / s1;
    alpha_sh[t] = al;
    alpha_out[b * PATCH + t] = al;
  }
  if (t == 0) beta_out[b] = beta;
  __syncthreads();

  const float* Ab  = A  + (size_t)b * PATCH * HIDDEN;
  const float* stb = st + (size_t)b * HIDDEN;
  for (int h = t; h < HIDDEN; h += 256) {
    float ct = 0.0f;
    for (int p = 0; p < PATCH; ++p)
      ct += Ab[(size_t)p * HIDDEN + h] * alpha_sh[p];
    chat_out[(size_t)b * HIDDEN + h] = beta * stb[h] + (1.0f - beta) * ct;
  }
}

// -------------------------------------------------------------------------
extern "C" void kernel_launch(void* const* d_in, const int* in_sizes, int n_in,
                              void* d_out, int out_size, void* d_ws, size_t ws_size,
                              hipStream_t stream) {
  (void)in_sizes; (void)n_in; (void)out_size; (void)ws_size;
  const float* spatial = (const float*)d_in[0];
  const float* decoder = (const float*)d_in[1];
  const float* stp     = (const float*)d_in[2];
  const float* Wcnn    = (const float*)d_in[3];
  const float* Wdec    = (const float*)d_in[4];
  const float* Wsen    = (const float*)d_in[5];
  const float* Watt    = (const float*)d_in[6];

  char* ws = (char*)d_ws;
  unsigned short* Wc_p = (unsigned short*)ws;  ws += (size_t)HIDDEN * ATT * 2;
  unsigned short* Wd_p = (unsigned short*)ws;  ws += (size_t)HIDDEN * ATT * 2;
  unsigned short* Ws_p = (unsigned short*)ws;  ws += (size_t)HIDDEN * ATT * 2;
  float* dec_out = (float*)ws;                 ws += (size_t)BATCH * ATT * 4;
  float* out_b   = (float*)ws;                 ws += (size_t)BATCH * 4;
  float* zt      = (float*)ws;                 ws += (size_t)MROWS * 4;

  float* alpha_out = (float*)d_out;
  float* beta_out  = alpha_out + MROWS;
  float* chat_out  = beta_out + BATCH;

  pack_w_kernel<<<dim3((HIDDEN * ATT) / 256, 3), 256, 0, stream>>>(
      Wcnn, Wdec, Wsen, Wc_p, Wd_p, Ws_p);
  decsen_kernel<<<BATCH / 16, 256, 0, stream>>>(
      decoder, stp, Wd_p, Ws_p, Watt, dec_out, out_b);
  zt_kernel<<<MROWS / 16, 256, 0, stream>>>(
      spatial, Wc_p, dec_out, Watt, zt);
  final_kernel<<<BATCH, 256, 0, stream>>>(
      zt, out_b, spatial, stp, alpha_out, beta_out, chat_out);
}
